// PCNN_20942260536125
// MI455X (gfx1250) — matrix operand order repaired and checked by reading, not compiled, not run on hardware
//
#include <hip/hip_runtime.h>
#include <hip/hip_bf16.h>

typedef __bf16 bf16_t;
typedef __attribute__((ext_vector_type(16))) __bf16 v16bf;
typedef __attribute__((ext_vector_type(8)))  __bf16 v8bf;
typedef __attribute__((ext_vector_type(8)))  float  v8f;

#define B_      1024
#define L_      256
#define D_      50
#define W_      230
#define KH_     10
#define CONVL_  247
#define DP_     52      // D+2
#define KROW_   64      // padded feature-row width (elements) = K per tap
#define KTOT_   640     // KH_*KROW_ (padded K)
#define NPAD_   240     // N padded to 15 tiles of 16
#define LABELS_ 53
#define HID3_   690     // 3*W_

// dynamic-LDS layout for conv_pool (no static __shared__ => base offset 0)
#define FEAT_ELEMS  ((L_ + 16) * KROW_)          // 17408 bf16
#define FEAT_BYTES  (FEAT_ELEMS * 2)             // 34816
#define WG_COLS     80                           // N columns staged per group
#define WLDS_BYTES  (WG_COLS * KTOT_ * 2)        // 102400
#define WLDS_OFF    FEAT_BYTES                   // 34816 (32B aligned)
#define UPOOL_OFF   (WLDS_OFF + WLDS_BYTES)      // 137216
#define SMEM_BYTES  (UPOOL_OFF + 3 * NPAD_ * 4)  // 140096

// order-preserving float <-> uint encoding so ds_max_u32 implements float max
__device__ __forceinline__ unsigned fenc(float f) {
    int b = __float_as_int(f);
    return (unsigned)(b ^ ((b >> 31) | 0x80000000));
}
__device__ __forceinline__ float fdec(unsigned u) {
    int b = (u & 0x80000000u) ? (int)(u ^ 0x80000000u) : ~(int)u;
    return __int_as_float(b);
}

// CDNA5 async copy: global -> LDS, 16B per lane, tracked by ASYNCcnt
__device__ __forceinline__ void async_g2l_b128(unsigned lds_byte_off, const void* gaddr) {
    asm volatile("global_load_async_to_lds_b128 %0, %1, off"
                 :: "v"(lds_byte_off),
                    "v"((unsigned long long)(size_t)gaddr)
                 : "memory");
}
__device__ __forceinline__ void wait_async0() {
    asm volatile("s_wait_asynccnt 0x0" ::: "memory");
}

// ---------------------------------------------------------------------------
// Kernel 0: repack conv weights fp32 [230][1][10][52] -> bf16 [240][640],
// K index = kh*64 + d (d 52..63 zero, n 230..239 zero); per-column K contiguous.
// ---------------------------------------------------------------------------
__global__ void prep_w(const float* __restrict__ cw, bf16_t* __restrict__ wb) {
    int i = blockIdx.x * 256 + threadIdx.x;
    if (i >= NPAD_ * KTOT_) return;
    int n = i / KTOT_, k = i - n * KTOT_;
    int kh = k >> 6, d = k & 63;
    float v = 0.f;
    if (n < W_ && d < DP_) v = cw[(n * KH_ + kh) * DP_ + d];
    wb[i] = (bf16_t)v;
}

// ---------------------------------------------------------------------------
// Kernel 1: fused embedding-gather + conv (implicit GEMM, bf16 WMMA, weights
// staged per-group into LDS via async-to-LDS DMA) + piecewise max-pool + tanh.
// One workgroup (8 waves) per batch row.
// ---------------------------------------------------------------------------
__global__ __launch_bounds__(256) void conv_pool(
    const int* __restrict__ x, const int* __restrict__ pos,
    const float* __restrict__ posEmb, const float* __restrict__ emb,
    const bf16_t* __restrict__ wb, const float* __restrict__ cb,
    float* __restrict__ hout)
{
    extern __shared__ char smem[];
    bf16_t*   feat  = (bf16_t*)smem;                    // [272][64] bf16
    bf16_t*   wlds  = (bf16_t*)(smem + WLDS_OFF);       // [80][640] bf16
    unsigned* upool = (unsigned*)(smem + UPOOL_OFF);    // [3][240] encoded f32

    const int b   = blockIdx.x;
    const int tid = threadIdx.x;

    // kick off async staging of weight group 0 (overlaps the gather below)
    {
        const char* src = (const char*)wb;              // group 0 base
        #pragma unroll
        for (int i = 0; i < 25; ++i) {                  // 6400 x 16B chunks / 256 thr
            const int c = tid + i * 256;
            async_g2l_b128(WLDS_OFF + c * 16, src + (long)c * 16);
        }
    }

    // init piecewise-max pools to encode(-1e30)
    const unsigned NEGK = fenc(-1e30f);
    for (int i = tid; i < 3 * NPAD_; i += 256) upool[i] = NEGK;

    // build bf16 feature tile in LDS: row t = [emb(x[b,t]) | posEmb | zeros]
    {
        const int t   = tid;                            // one row per thread
        const int tok = x[b * L_ + t];
        const float2* e2 = (const float2*)(emb + (long)tok * D_); // 200B rows, 8B aligned
        bf16_t* row = &feat[t * KROW_];
        #pragma unroll
        for (int i = 0; i < 25; ++i) {
            float2 v = e2[i];
            row[2 * i]     = (bf16_t)v.x;
            row[2 * i + 1] = (bf16_t)v.y;
        }
        const float* pe = posEmb + ((long)b * L_ + t) * 2;
        row[50] = (bf16_t)pe[0];
        row[51] = (bf16_t)pe[1];
        #pragma unroll
        for (int d = DP_; d < KROW_; ++d) row[d] = (bf16_t)0.f;
        bf16_t* zr = &feat[L_ * KROW_];                 // zero pad rows 256..271
        #pragma unroll
        for (int j = 0; j < 4; ++j) zr[tid * 4 + j] = (bf16_t)0.f;
    }
    wait_async0();
    __syncthreads();

    const int wave  = tid >> 5;
    const int lane  = tid & 31;
    const int lrow  = lane & 15;   // M row (A) or N col (B/C) within tile
    const int lhalf = lane >> 4;   // K-half selector (this code's contiguous k = 16*lhalf + s)
    const int p1 = pos[b * 2 + 0];
    const int p2 = pos[b * 2 + 1];

    for (int ntg = 0; ntg < 3; ++ntg) {                 // 3 groups x 5 N-tiles
        // lane's B column base inside the staged group, K contiguous
        const bf16_t* wcol = wlds + (long)lrow * KTOT_ + lhalf * 16;

        for (int mi = 0; mi < 2; ++mi) {
            const int mt   = wave + mi * 8;             // 16 M-tiles over t=0..255
            const int trow = mt * 16 + lrow;
            v8f acc[5] = {};
            for (int ks = 0; ks < 20; ++ks) {           // K = 640 = 20 steps of 32
                const int kh   = ks >> 1;
                const int aoff = (trow + kh) * KROW_ + ((ks & 1) << 5) + lhalf * 16;
                // A 16x32 bf16 fragment: lane half holds K{0..15}/{16..31}
                v8bf alo = *(const v8bf*)&feat[aoff];
                v8bf ahi = *(const v8bf*)&feat[aoff + 8];
                v16bf amat = __builtin_shufflevector(alo, ahi,
                    0, 1, 2, 3, 4, 5, 6, 7, 8, 9, 10, 11, 12, 13, 14, 15);
                #pragma unroll
                for (int q = 0; q < 5; ++q) {           // reuse A across 5 N-tiles
                    v16bf bmat = *(const v16bf*)(wcol + (long)q * 16 * KTOT_ + ks * 32);
                    acc[q] = __builtin_amdgcn_wmma_f32_16x16x32_bf16(
                        false, amat, false, bmat, (short)0, acc[q], false, false);
                }
            }
            // fused piecewise max-pool: C layout M = r + 8*lhalf, N = lrow
            #pragma unroll
            for (int q = 0; q < 5; ++q) {
                const int ncol = (ntg * 5 + q) * 16 + lrow;
                if (ncol < W_) {
                    const float bias = cb[ncol];
                    float s0 = -1e30f, s1 = -1e30f, s2 = -1e30f;
                    #pragma unroll
                    for (int r = 0; r < 8; ++r) {
                        const int t = mt * 16 + r + lhalf * 8;
                        if (t < CONVL_) {
                            const float v = acc[q][r] + bias;
                            if (t <= p1)      s0 = fmaxf(s0, v);   // [0, p1]
                            else if (t <= p2) s1 = fmaxf(s1, v);   // (p1, p2]
                            else              s2 = fmaxf(s2, v);   // (p2, end)
                        }
                    }
                    if (s0 > -1e30f) atomicMax(&upool[0 * NPAD_ + ncol], fenc(s0));
                    if (s1 > -1e30f) atomicMax(&upool[1 * NPAD_ + ncol], fenc(s1));
                    if (s2 > -1e30f) atomicMax(&upool[2 * NPAD_ + ncol], fenc(s2));
                }
            }
        }
        __syncthreads();                                // all waves done with wlds
        if (ntg < 2) {                                  // restage next weight group
            const char* src = (const char*)(wb + (long)(ntg + 1) * WG_COLS * KTOT_);
            #pragma unroll
            for (int i = 0; i < 25; ++i) {
                const int c = tid + i * 256;
                async_g2l_b128(WLDS_OFF + c * 16, src + (long)c * 16);
            }
            wait_async0();
            __syncthreads();
        }
    }

    if (tid < W_) {
        #pragma unroll
        for (int s = 0; s < 3; ++s) {
            float v = fdec(upool[s * NPAD_ + tid]);
            hout[(long)b * HID3_ + s * W_ + tid] = tanhf(v);
        }
    }
}

// ---------------------------------------------------------------------------
// Kernel 2: dense head. Linear(690->230)+BN+PReLU, Linear(230->230)+BN+PReLU,
// Linear(230->53), softmax. One block per batch row (<1% of total FLOPs).
// ---------------------------------------------------------------------------
__global__ __launch_bounds__(256) void dense_head(
    const float* __restrict__ hin,
    const float* __restrict__ d0w, const float* __restrict__ d0b,
    const float* __restrict__ g0, const float* __restrict__ be0,
    const float* __restrict__ m0, const float* __restrict__ v0,
    const float* __restrict__ a0,
    const float* __restrict__ d1w, const float* __restrict__ d1b,
    const float* __restrict__ g1, const float* __restrict__ be1,
    const float* __restrict__ m1, const float* __restrict__ v1,
    const float* __restrict__ a1,
    const float* __restrict__ ow, const float* __restrict__ ob,
    float* __restrict__ out)
{
    __shared__ float hb[HID3_];
    __shared__ float h0[W_];
    __shared__ float h1[W_];
    __shared__ float lg[LABELS_];
    __shared__ float red[2];
    const int b = blockIdx.x, tid = threadIdx.x;

    for (int j = tid; j < HID3_; j += 256) hb[j] = hin[(long)b * HID3_ + j];
    __syncthreads();
    if (tid < W_) {
        float acc = d0b[tid];
        const float* wr = d0w + (long)tid * HID3_;
        for (int j = 0; j < HID3_; ++j) acc = fmaf(hb[j], wr[j], acc);
        float y = (acc - m0[tid]) * (g0[tid] * rsqrtf(v0[tid] + 1e-5f)) + be0[tid];
        h0[tid] = y > 0.f ? y : a0[0] * y;
    }
    __syncthreads();
    if (tid < W_) {
        float acc = d1b[tid];
        const float* wr = d1w + (long)tid * W_;
        for (int j = 0; j < W_; ++j) acc = fmaf(h0[j], wr[j], acc);
        float y = (acc - m1[tid]) * (g1[tid] * rsqrtf(v1[tid] + 1e-5f)) + be1[tid];
        h1[tid] = y > 0.f ? y : a1[0] * y;
    }
    __syncthreads();
    if (tid < LABELS_) {
        float acc = ob[tid];
        const float* wr = ow + (long)tid * W_;
        for (int j = 0; j < W_; ++j) acc = fmaf(h1[j], wr[j], acc);
        lg[tid] = acc;
    }
    __syncthreads();
    if (tid == 0) {
        float mx = lg[0];
        for (int l = 1; l < LABELS_; ++l) mx = fmaxf(mx, lg[l]);
        float sm = 0.f;
        for (int l = 0; l < LABELS_; ++l) sm += __expf(lg[l] - mx);
        red[0] = mx; red[1] = sm;
    }
    __syncthreads();
    if (tid < LABELS_)
        out[(long)b * LABELS_ + tid] = __expf(lg[tid] - red[0]) / red[1];
}

// ---------------------------------------------------------------------------
extern "C" void kernel_launch(void* const* d_in, const int* in_sizes, int n_in,
                              void* d_out, int out_size, void* d_ws, size_t ws_size,
                              hipStream_t stream)
{
    const int*   x   = (const int*)d_in[0];
    const int*   pos = (const int*)d_in[1];
    const float* pe  = (const float*)d_in[2];
    const float* emb = (const float*)d_in[3];
    const float* cw  = (const float*)d_in[4];
    const float* cb  = (const float*)d_in[5];
    const float* d0w = (const float*)d_in[6];
    const float* d0b = (const float*)d_in[7];
    const float* g0  = (const float*)d_in[8];
    const float* be0 = (const float*)d_in[9];
    const float* m0  = (const float*)d_in[10];
    const float* v0  = (const float*)d_in[11];
    const float* a0  = (const float*)d_in[12];
    const float* d1w = (const float*)d_in[13];
    const float* d1b = (const float*)d_in[14];
    const float* g1  = (const float*)d_in[15];
    const float* be1 = (const float*)d_in[16];
    const float* m1  = (const float*)d_in[17];
    const float* v1  = (const float*)d_in[18];
    const float* a1  = (const float*)d_in[19];
    const float* ow  = (const float*)d_in[20];
    const float* ob  = (const float*)d_in[21];
    float* out = (float*)d_out;

    // workspace: bf16 weights [240*640] at 0 (307200 B), pooled h [1024*690] f32 after
    bf16_t* wbw  = (bf16_t*)d_ws;
    float*  hbuf = (float*)((char*)d_ws + 327680);

    prep_w<<<(NPAD_ * KTOT_ + 255) / 256, 256, 0, stream>>>(cw, wbw);
    conv_pool<<<B_, 256, SMEM_BYTES, stream>>>(x, pos, pe, emb, wbw, cb, hbuf);
    dense_head<<<B_, 256, 0, stream>>>(hbuf, d0w, d0b, g0, be0, m0, v0, a0,
                                       d1w, d1b, g1, be1, m1, v1, a1, ow, ob, out);
}
